// GroupedQueryAttention_10204842295767
// MI455X (gfx1250) — matrix operand order repaired and checked by reading, not compiled, not run on hardware
//
#include <hip/hip_runtime.h>
#include <hip/hip_bf16.h>

// ---------------------------------------------------------------------------
// GQA forward for MI455X (gfx1250): all matmuls via v_wmma_f32_16x16x32_f16.
// ---------------------------------------------------------------------------

typedef _Float16 h16v __attribute__((ext_vector_type(16)));
typedef _Float16 h8v  __attribute__((ext_vector_type(8)));
typedef _Float16 h2v  __attribute__((ext_vector_type(2)));
typedef float    f8v  __attribute__((ext_vector_type(8)));

#define B_   2
#define T_   2048
#define DIM_ 2048
#define HQ_  32
#define HKV_ 8
#define DH_  64

static __device__ __forceinline__ h16v cat8(h8v lo, h8v hi) {
  return __builtin_shufflevector(lo, hi, 0,1,2,3,4,5,6,7,8,9,10,11,12,13,14,15);
}

static __device__ __forceinline__ f8v wmma_f16(h16v a, h16v b, f8v c) {
  // D = A(16x32 f16) * B(32x16 f16) + C(16x16 f32)
  return __builtin_amdgcn_wmma_f32_16x16x32_f16(false, a, false, b, (short)0, c,
                                                false, false);
}

// packed 2xf16 max -> v_pk_max_num_f16
static __device__ __forceinline__ int pkmax2(int a, int b) {
  h2v x = __builtin_bit_cast(h2v, a);
  h2v y = __builtin_bit_cast(h2v, b);
#if __has_builtin(__builtin_elementwise_max)
  h2v r = __builtin_elementwise_max(x, y);
#else
  h2v r;
  r[0] = (x[0] > y[0]) ? x[0] : y[0];
  r[1] = (x[1] > y[1]) ? x[1] : y[1];
#endif
  return __builtin_bit_cast(int, r);
}

// ---------------------------------------------------------------------------
// Generic GEMM: C[M,N] = A[M,K] * W[K,N].  A: f32 or f16; W: f32 (converted to
// f16 in LDS); f32 WMMA accumulate; output f16 or f32.
// 128x128x32 block tile, 256 threads = 8 waves, each wave a 32x64 sub-tile.
// Software-pipelined double buffering: global->reg loads for tile k+1 overlap
// the WMMAs on tile k; one barrier per K-step.
// ---------------------------------------------------------------------------
template <typename AT, typename OT>
__global__ __launch_bounds__(256) void gemm_wmma_kernel(
    const AT* __restrict__ A, const float* __restrict__ W, OT* __restrict__ C,
    int M, int N, int K) {
  constexpr int BM = 128, BN = 128, BK = 32;
  constexpr int SA = BK + 8;  // 40 halves = 80B row stride (16B aligned)
  __shared__ alignas(16) _Float16 Alds[2][BM][SA];
  __shared__ alignas(16) _Float16 Blds[2][BN][SA];  // transposed: [n][k]

  const int tid  = threadIdx.x;
  const int lane = tid & 31;
  const int wid  = tid >> 5;
  const int g = lane >> 4, ln = lane & 15;

  const int m0 = blockIdx.y * BM;
  const int n0 = blockIdx.x * BN;
  const int moff = (wid >> 1) * 32;  // 4 wave-rows
  const int noff = (wid & 1) * 64;   // 2 wave-cols

  f8v acc[2][4] = {};

  const int arow = tid >> 1, ac0 = (tid & 1) * 16;  // 128x32 A tile fill
  const int wrow = tid >> 3, wc0 = (tid & 7) * 16;  // 32x128 W tile fill

  AT    areg[16];
  float wreg[16];

  auto gload = [&](int k0) {
    const AT* asrc = A + (size_t)(m0 + arow) * K + (k0 + ac0);
#pragma unroll
    for (int i = 0; i < 16; ++i) areg[i] = asrc[i];
    const float* wsrc = W + (size_t)(k0 + wrow) * N + (n0 + wc0);
#pragma unroll
    for (int i = 0; i < 16; ++i) wreg[i] = wsrc[i];
  };
  auto lstore = [&](int buf) {
#pragma unroll
    for (int i = 0; i < 16; ++i) Alds[buf][arow][ac0 + i] = (_Float16)areg[i];
#pragma unroll
    for (int i = 0; i < 16; ++i) Blds[buf][wc0 + i][wrow] = (_Float16)wreg[i];
  };

  gload(0);
  lstore(0);
  int buf = 0;

  for (int k0 = 0; k0 < K; k0 += BK) {
    const bool more = (k0 + BK) < K;
    if (more) gload(k0 + BK);  // in flight across this step's WMMAs
    __syncthreads();           // LDS[buf] ready

    h16v afr[2], bfr[4];
#pragma unroll
    for (int mt = 0; mt < 2; ++mt) {  // A frag: lane(g,m): k = g*8+i / 16+g*8+i
      const _Float16* p = &Alds[buf][moff + mt * 16 + ln][0];
      afr[mt] = cat8(*(const h8v*)(p + g * 8), *(const h8v*)(p + 16 + g * 8));
    }
#pragma unroll
    for (int nt = 0; nt < 4; ++nt) {  // B frag: lane(g,n): k = g*8+i / 16+g*8+i
      const _Float16* p = &Blds[buf][noff + nt * 16 + ln][0];
      bfr[nt] = cat8(*(const h8v*)(p + g * 8), *(const h8v*)(p + 16 + g * 8));
    }
#pragma unroll
    for (int mt = 0; mt < 2; ++mt)
#pragma unroll
      for (int nt = 0; nt < 4; ++nt)
        acc[mt][nt] = wmma_f16(afr[mt], bfr[nt], acc[mt][nt]);

    if (more) lstore(buf ^ 1);  // safe: peers read only LDS[buf] this step
    buf ^= 1;
  }

  // epilogue: C frag reg r -> row r+8g, col = ln
#pragma unroll
  for (int mt = 0; mt < 2; ++mt)
#pragma unroll
    for (int nt = 0; nt < 4; ++nt)
#pragma unroll
      for (int r = 0; r < 8; ++r) {
        const int row = m0 + moff + mt * 16 + r + 8 * g;
        const int col = n0 + noff + nt * 16 + ln;
        C[(size_t)row * N + col] = (OT)acc[mt][nt][r];
      }
}

// ---------------------------------------------------------------------------
// Flash GQA: grid(B*H, T/64), 128 threads (4 waves); wave w owns q rows
// [qb*64 + w*16, +16).  One softmax pass per 64-kv block (8 S-WMMAs), causal
// mask (== reference's additive -1e9 mask) only on the diagonal block.
// Row max: packed-f16 batched xor-shuffle reduction (shift-invariance of
// softmax makes the f16-rounded max a valid stabilizing shift).
// Row sums: WMMA against an all-ones B fragment.  O += P*V via WMMA.
// Q pre-scaled by 1/sqrt(64)=0.125 (exact power of two in f16).
// Next kv block prefetched (global_prefetch_b8) while current block computes.
// ---------------------------------------------------------------------------
__global__ __launch_bounds__(128) void gqa_flash_kernel(
    const _Float16* __restrict__ Q, const _Float16* __restrict__ K,
    const _Float16* __restrict__ V, _Float16* __restrict__ O) {
  __shared__ alignas(16) _Float16 Qs[64][72];
  __shared__ alignas(16) _Float16 Ks[64][72];
  __shared__ alignas(16) _Float16 Vt[64][72];     // V transposed: [d][kv]
  __shared__ alignas(16) _Float16 Ps[4][16][72];  // per-wave P repack (16x64)

  const int bh = blockIdx.x;
  const int b  = bh / HQ_;
  const int h  = bh % HQ_;
  const int hk = h / (HQ_ / HKV_);
  const int qb = blockIdx.y;

  const int tid  = threadIdx.x;
  const int wid  = tid >> 5;
  const int lane = tid & 31;
  const int g = lane >> 4, ln = lane & 15;

  const _Float16* Qg = Q + (size_t)(b * T_ + qb * 64) * (HQ_ * DH_) + h * DH_;
  const _Float16* Kg = K + (size_t)(b * T_) * (HKV_ * DH_) + hk * DH_;
  const _Float16* Vg = V + (size_t)(b * T_) * (HKV_ * DH_) + hk * DH_;

  const int srow = tid >> 1, sc0 = (tid & 1) * 32;  // staging assignment

  {  // load 64x64 Q tile, pre-scaled by 0.125 (exact power-of-two in f16)
    const _Float16* src = Qg + (size_t)srow * (HQ_ * DH_) + sc0;
#pragma unroll
    for (int i = 0; i < 32; ++i) Qs[srow][sc0 + i] = src[i] * (_Float16)0.125f;
  }
  // prefetch first K/V block while Q tile settles
  __builtin_prefetch(Kg + (size_t)srow * (HKV_ * DH_) + sc0, 0, 1);
  __builtin_prefetch(Vg + (size_t)srow * (HKV_ * DH_) + sc0, 0, 1);
  __syncthreads();

  h16v qa[2];  // Q A-fragments, contraction d=0..31 / 32..63
#pragma unroll
  for (int dc = 0; dc < 2; ++dc) {
    const _Float16* p = &Qs[wid * 16 + ln][dc * 32];
    qa[dc] = cat8(*(const h8v*)(p + g * 8), *(const h8v*)(p + 16 + g * 8));
  }

  h16v vone;  // all-ones B fragment for WMMA row sums
#pragma unroll
  for (int i = 0; i < 16; ++i) vone[i] = (_Float16)1.0f;

  float mrow[8], lrow[8];
  f8v o[4] = {};
#pragma unroll
  for (int r = 0; r < 8; ++r) { mrow[r] = -3.0e38f; lrow[r] = 0.0f; }

  const int qbase = qb * 64 + wid * 16;

  // One online-softmax pass over a full 64-kv block.
  auto process_block = [&](int kvb, bool masked) {
    f8v s[4] = {};
#pragma unroll
    for (int j = 0; j < 4; ++j) {  // S = (Q/8) * K^T, four 16-col subtiles
#pragma unroll
      for (int dc = 0; dc < 2; ++dc) {
        const _Float16* p = &Ks[j * 16 + ln][dc * 32];
        h16v bf = cat8(*(const h8v*)(p + g * 8), *(const h8v*)(p + 16 + g * 8));
        s[j] = wmma_f16(qa[dc], bf, s[j]);
      }
    }
    // mask (diagonal block only) + per-row max over the 4 subtiles
    float val[8];
#pragma unroll
    for (int r = 0; r < 8; ++r) {
      const int qi = qbase + r + 8 * g;
      if (masked) {
#pragma unroll
        for (int j = 0; j < 4; ++j)
          if (kvb + j * 16 + ln > qi) s[j][r] = -1.0e9f;
      }
      val[r] = fmaxf(fmaxf(s[0][r], s[1][r]), fmaxf(s[2][r], s[3][r]));
    }
    // packed-f16, step-major xor reduction: 4 independent bpermutes per step
    int pk[4];
#pragma unroll
    for (int t = 0; t < 4; ++t) {
      h2v p;
      p[0] = (_Float16)val[2 * t];
      p[1] = (_Float16)val[2 * t + 1];
      pk[t] = __builtin_bit_cast(int, p);
    }
#pragma unroll
    for (int m = 1; m < 16; m <<= 1) {
      int tmp[4];
#pragma unroll
      for (int t = 0; t < 4; ++t) tmp[t] = __shfl_xor(pk[t], m, 32);
#pragma unroll
      for (int t = 0; t < 4; ++t) pk[t] = pkmax2(pk[t], tmp[t]);
    }
    float alph[8];
#pragma unroll
    for (int r = 0; r < 8; ++r) {
      const h2v pm = __builtin_bit_cast(h2v, pk[r >> 1]);
      const float rowm  = (float)pm[r & 1];
      const float mnew  = fmaxf(mrow[r], rowm);
      const float alpha = __expf(mrow[r] - mnew);
      mrow[r] = mnew;
      alph[r] = alpha;
      o[0][r] *= alpha; o[1][r] *= alpha; o[2][r] *= alpha; o[3][r] *= alpha;
#pragma unroll
      for (int j = 0; j < 4; ++j)  // repack P to A-frag row layout
        Ps[wid][r + 8 * g][j * 16 + ln] = (_Float16)__expf(s[j][r] - mnew);
    }
    __builtin_amdgcn_wave_barrier();  // same-wave LDS is in-order; pin sched
    const _Float16* pp = &Ps[wid][ln][0];
    h16v pa0 = cat8(*(const h8v*)(pp + g * 8),      *(const h8v*)(pp + 16 + g * 8));
    h16v pa1 = cat8(*(const h8v*)(pp + 32 + g * 8), *(const h8v*)(pp + 48 + g * 8));
    f8v zero = {};
    f8v sums = wmma_f16(pa0, vone, zero);  // row sums of P, replicated per col
    sums = wmma_f16(pa1, vone, sums);
#pragma unroll
    for (int nt = 0; nt < 4; ++nt) {  // O += P(16x64) * V(64x16-per-tile)
      const _Float16* vp = &Vt[nt * 16 + ln][0];
      h16v vb0 = cat8(*(const h8v*)(vp + g * 8),      *(const h8v*)(vp + 16 + g * 8));
      h16v vb1 = cat8(*(const h8v*)(vp + 32 + g * 8), *(const h8v*)(vp + 48 + g * 8));
      o[nt] = wmma_f16(pa0, vb0, o[nt]);
      o[nt] = wmma_f16(pa1, vb1, o[nt]);
    }
#pragma unroll
    for (int r = 0; r < 8; ++r) lrow[r] = lrow[r] * alph[r] + sums[r];
    __builtin_amdgcn_wave_barrier();
  };

  for (int kb = 0; kb <= qb; ++kb) {
    __syncthreads();
    {  // stage K row-major, V transposed
      const _Float16* ks = Kg + (size_t)(kb * 64 + srow) * (HKV_ * DH_) + sc0;
      const _Float16* vs = Vg + (size_t)(kb * 64 + srow) * (HKV_ * DH_) + sc0;
#pragma unroll
      for (int i = 0; i < 32; ++i) Ks[srow][sc0 + i] = ks[i];
#pragma unroll
      for (int i = 0; i < 32; ++i) Vt[sc0 + i][srow] = vs[i];
      if (kb < qb) {  // warm next block's tile while this one computes
        __builtin_prefetch(ks + 64 * (HKV_ * DH_), 0, 1);
        __builtin_prefetch(vs + 64 * (HKV_ * DH_), 0, 1);
      }
    }
    __syncthreads();

    if (kb < qb) process_block(kb * 64, false);  // interior: unmasked
    else         process_block(kb * 64, true);   // diagonal: causal mask
  }

  // normalize and store f16 attn output [B*T, H*Dh]
#pragma unroll
  for (int r = 0; r < 8; ++r) {
    const float inv = 1.0f / lrow[r];
    const int qi = qbase + r + 8 * g;
    _Float16* dst = O + (size_t)(b * T_ + qi) * (HQ_ * DH_) + h * DH_;
#pragma unroll
    for (int nt = 0; nt < 4; ++nt)
      dst[nt * 16 + ln] = (_Float16)(o[nt][r] * inv);
  }
}

// ---------------------------------------------------------------------------
extern "C" void kernel_launch(void* const* d_in, const int* in_sizes, int n_in,
                              void* d_out, int out_size, void* d_ws,
                              size_t ws_size, hipStream_t stream) {
  (void)in_sizes; (void)n_in; (void)out_size; (void)ws_size;
  const float* x  = (const float*)d_in[0];
  const float* wq = (const float*)d_in[1];
  const float* wk = (const float*)d_in[2];
  const float* wv = (const float*)d_in[3];
  const float* wo = (const float*)d_in[4];
  // d_in[5] = attention_mask: causal, applied analytically in-kernel.

  const int M = B_ * T_;  // 4096
  _Float16* Qb = (_Float16*)d_ws;                 // 4096 x 2048 f16 (16 MB)
  _Float16* Kb = Qb + (size_t)M * (HQ_ * DH_);    // 4096 x 512  f16 ( 4 MB)
  _Float16* Vb = Kb + (size_t)M * (HKV_ * DH_);   // 4096 x 512  f16 ( 4 MB)
  _Float16* Ab = Vb + (size_t)M * (HKV_ * DH_);   // 4096 x 2048 f16 (16 MB)

  const dim3 blk(256);
  gemm_wmma_kernel<float, _Float16>
      <<<dim3(DIM_ / 128, M / 128), blk, 0, stream>>>(x, wq, Qb, M, HQ_ * DH_, DIM_);
  gemm_wmma_kernel<float, _Float16>
      <<<dim3((HKV_ * DH_) / 128, M / 128), blk, 0, stream>>>(x, wk, Kb, M, HKV_ * DH_, DIM_);
  gemm_wmma_kernel<float, _Float16>
      <<<dim3((HKV_ * DH_) / 128, M / 128), blk, 0, stream>>>(x, wv, Vb, M, HKV_ * DH_, DIM_);

  gqa_flash_kernel<<<dim3(B_ * HQ_, T_ / 64), dim3(128), 0, stream>>>(Qb, Kb, Vb, Ab);

  gemm_wmma_kernel<_Float16, float>
      <<<dim3(DIM_ / 128, M / 128), blk, 0, stream>>>(Ab, wo, (float*)d_out, M, DIM_, DIM_);
}
